// GCL_32341103739238
// MI455X (gfx1250) — compile-verified, matrix operand order checked by reading, not run on hardware
//
#include <hip/hip_runtime.h>
#include <hip/hip_fp16.h>
#include <hip/hip_bf16.h>

typedef __attribute__((ext_vector_type(16))) _Float16 v16h;
typedef __attribute__((ext_vector_type(8)))  float    v8f;

#define DIMS 64
#define T_INV 2.0f   // 1/T, T=0.5
#define NT 4         // 16-col tiles per wave in the SSL GEMM

// ---------------------------------------------------------------------------
// hardware float atomic (global_atomic_add_f32), no CAS loop
__device__ __forceinline__ void atomAddF(float* p, float v) {
    unsafeAtomicAdd(p, v);
}

// ---------------------------------------------------------------------------
// e_cur = emb; acc = emb
__global__ __launch_bounds__(256) void init_copy_kernel(const float* __restrict__ emb,
                                                        float* __restrict__ e_cur,
                                                        float* __restrict__ acc, int nd) {
    int i = blockIdx.x * 256 + threadIdx.x;
    if (i < nd) {
        float v = emb[i];
        e_cur[i] = v;
        acc[i]   = v;
    }
}

// acc += e_next
__global__ __launch_bounds__(256) void acc_add_kernel(float* __restrict__ acc,
                                                      const float* __restrict__ e, int nd) {
    int i = blockIdx.x * 256 + threadIdx.x;
    if (i < nd) acc[i] += e[i];
}

// ---------------------------------------------------------------------------
// COO SpMM: e_next[row] += val * e_cur[col].
// 16 lanes per edge, float4/lane -> b128 loads, 4 hw f32 atomics per lane.
__global__ __launch_bounds__(256) void spmm_kernel(const float* __restrict__ e_cur,
                                                   float* __restrict__ e_next,
                                                   const int* __restrict__ erow,
                                                   const int* __restrict__ ecol,
                                                   const float* __restrict__ vals, int E) {
    int e = blockIdx.x * 16 + (threadIdx.x >> 4);
    if (e >= E) return;
    int l = threadIdx.x & 15;
    int r = erow[e];
    int c = ecol[e];
    float v = vals[e];
    float4 x = ((const float4*)(e_cur + (size_t)c * DIMS))[l];
    float* dst = e_next + (size_t)r * DIMS + l * 4;
    atomAddF(dst,     v * x.x);
    atomAddF(dst + 1, v * x.y);
    atomAddF(dst + 2, v * x.z);
    atomAddF(dst + 3, v * x.w);
}

// ---------------------------------------------------------------------------
// gather acc[nodes[b]] -> L1 normalize -> f32 row (for pos score) + f16 row (for WMMA)
__global__ __launch_bounds__(256) void gather_l1norm_kernel(const float* __restrict__ acc,
                                                            const int* __restrict__ nodes,
                                                            float* __restrict__ nf,
                                                            _Float16* __restrict__ nh) {
    int b = blockIdx.x * 8 + (threadIdx.x >> 5);
    int lane = threadIdx.x & 31;
    int node = nodes[b];
    float2 x = ((const float2*)(acc + (size_t)node * DIMS))[lane];
    float s = fabsf(x.x) + fabsf(x.y);
#pragma unroll
    for (int m = 16; m >= 1; m >>= 1) s += __shfl_xor(s, m, 32);
    float inv = 1.0f / fmaxf(s, 1e-12f);
    float2 y; y.x = x.x * inv; y.y = x.y * inv;
    ((float2*)(nf + (size_t)b * DIMS))[lane] = y;
    _Float16* dh = nh + (size_t)b * DIMS + lane * 2;
    dh[0] = (_Float16)y.x;
    dh[1] = (_Float16)y.y;
}

// ---------------------------------------------------------------------------
// ISA 16-bit A/B 16x32 fragment: pair i -> k = 32*step + 16*(i>>2) + 8*half + 2*(i&3)
__device__ __forceinline__ v16h load_frag(const _Float16* __restrict__ row,
                                          int half, int step) {
    v16h f;
#pragma unroll
    for (int i = 0; i < 8; ++i) {
        int k = (step << 5) + ((i >> 2) << 4) + (half << 3) + ((i & 3) << 1);
        f[2 * i]     = row[k];
        f[2 * i + 1] = row[k + 1];
    }
    return f;
}

// Fused ttl = n1 @ n2^T with rowsum[b] += sum_n exp(ttl[b,n]/T).
// Wave owns one 16-row tile x NT 16-col tiles; A fragments reused across NT tiles,
// exp accumulated per-lane in registers; one reduce + atomic per row fragment at end.
// Block = 8 waves -> 16 rows x (128*NT) cols. Grid (B/(128*NT), B/16). EXEC all-1s.
__global__ __launch_bounds__(256) void ssl_gemm_kernel(const _Float16* __restrict__ Ah,
                                                       const _Float16* __restrict__ Bh,
                                                       float* __restrict__ rowsum) {
    const int lane = threadIdx.x & 31;
    const int wave = threadIdx.x >> 5;
    const int half = lane >> 4;
    const int m0 = blockIdx.y << 4;
    const int n0 = (blockIdx.x * 8 + wave) * (16 * NT);

    const _Float16* arow = Ah + (size_t)(m0 + (lane & 15)) * DIMS;
    const v16h a0 = load_frag(arow, half, 0);
    const v16h a1 = load_frag(arow, half, 1);

    v8f accv = {};
#pragma unroll
    for (int t = 0; t < NT; ++t) {
        const _Float16* brow = Bh + (size_t)(n0 + t * 16 + (lane & 15)) * DIMS;
        const v16h b0 = load_frag(brow, half, 0);
        const v16h b1 = load_frag(brow, half, 1);
        v8f c = {};
        c = __builtin_amdgcn_wmma_f32_16x16x32_f16(false, a0, false, b0,
                                                   (short)0, c, false, false);
        c = __builtin_amdgcn_wmma_f32_16x16x32_f16(false, a1, false, b1,
                                                   (short)0, c, false, false);
#pragma unroll
        for (int r = 0; r < 8; ++r)
            accv[r] += __expf(c[r] * T_INV);
    }

    // C layout: vgpr r, lane l -> row m0 + r + 8*half, col (l & 15).
    // xor masks < 16 never cross the 16-lane half, so width-32 shuffles are safe
    // and avoid the width-16 clamp sequence.
#pragma unroll
    for (int r = 0; r < 8; ++r) {
        float v = accv[r];
        v += __shfl_xor(v, 1, 32);
        v += __shfl_xor(v, 2, 32);
        v += __shfl_xor(v, 4, 32);
        v += __shfl_xor(v, 8, 32);
        if ((lane & 15) == 0)
            atomAddF(&rowsum[m0 + r + (half << 3)], v);
    }
}

// ---------------------------------------------------------------------------
// ssl partial: sum_b ( log(rowsum[b]) - dot(n1[b],n2[b])/T )  -> accum[0]
__global__ __launch_bounds__(256) void ssl_pos_kernel(const float* __restrict__ n1f,
                                                      const float* __restrict__ n2f,
                                                      const float* __restrict__ rowsum,
                                                      float* __restrict__ accum) {
    int b = blockIdx.x * 8 + (threadIdx.x >> 5);
    int lane = threadIdx.x & 31;
    float2 x = ((const float2*)(n1f + (size_t)b * DIMS))[lane];
    float2 y = ((const float2*)(n2f + (size_t)b * DIMS))[lane];
    float d = x.x * y.x + x.y * y.y;
#pragma unroll
    for (int m = 16; m >= 1; m >>= 1) d += __shfl_xor(d, m, 32);
    if (lane == 0) {
        float partial = logf(rowsum[b]) - d * T_INV;
        atomicAdd(&accum[0], partial);
    }
}

// ---------------------------------------------------------------------------
// BPR + reg. acc0 holds 4*emb_all (layer-mean skipped) -> dot scale 1/16.
__global__ __launch_bounds__(256) void bpr_kernel(const float* __restrict__ acc0,
                                                  const float* __restrict__ emb,
                                                  const int* __restrict__ nl,
                                                  const int* __restrict__ pl,
                                                  const int* __restrict__ ngl,
                                                  float* __restrict__ accum) {
    __shared__ float s_b[8];
    __shared__ float s_r[8];
    int p = blockIdx.x * 8 + (threadIdx.x >> 5);
    int lane = threadIdx.x & 31;
    int iu = nl[p], iv = pl[p], ig = ngl[p];
    float2 u  = ((const float2*)(acc0 + (size_t)iu * DIMS))[lane];
    float2 v  = ((const float2*)(acc0 + (size_t)iv * DIMS))[lane];
    float2 g  = ((const float2*)(acc0 + (size_t)ig * DIMS))[lane];
    float2 u0 = ((const float2*)(emb  + (size_t)iu * DIMS))[lane];
    float2 v0 = ((const float2*)(emb  + (size_t)iv * DIMS))[lane];
    float2 g0 = ((const float2*)(emb  + (size_t)ig * DIMS))[lane];
    float pd = u.x * v.x + u.y * v.y;
    float nd = u.x * g.x + u.y * g.y;
    float rg = u0.x * u0.x + u0.y * u0.y + v0.x * v0.x + v0.y * v0.y +
               g0.x * g0.x + g0.y * g0.y;
#pragma unroll
    for (int m = 16; m >= 1; m >>= 1) {
        pd += __shfl_xor(pd, m, 32);
        nd += __shfl_xor(nd, m, 32);
        rg += __shfl_xor(rg, m, 32);
    }
    if (lane == 0) {
        float x = (nd - pd) * 0.0625f;                       // (1/4)^2 layer-mean scale
        s_b[threadIdx.x >> 5] = (x > 20.0f) ? x : log1pf(__expf(x));
        s_r[threadIdx.x >> 5] = rg;
    }
    __syncthreads();
    if (threadIdx.x == 0) {
        float sb = 0.0f, sr = 0.0f;
#pragma unroll
        for (int i = 0; i < 8; ++i) { sb += s_b[i]; sr += s_r[i]; }
        atomicAdd(&accum[1], sb);
        atomicAdd(&accum[2], sr);
    }
}

// out = ssl + bpr_mean + 1e-4 * (0.5 * reg_sum / B)
__global__ void finalize_kernel(const float* __restrict__ accum, float* __restrict__ out,
                                float invP, float halfOverB) {
    out[0] = accum[0] + accum[1] * invP + 1e-4f * (halfOverB * accum[2]);
}

// ---------------------------------------------------------------------------
extern "C" void kernel_launch(void* const* d_in, const int* in_sizes, int n_in,
                              void* d_out, int out_size, void* d_ws, size_t ws_size,
                              hipStream_t stream) {
    const float* emb   = (const float*)d_in[0];
    const float* vals1 = (const float*)d_in[1];
    const float* vals2 = (const float*)d_in[2];
    const float* vals0 = (const float*)d_in[3];
    const int*   erow  = (const int*)d_in[4];
    const int*   ecol  = (const int*)d_in[5];
    const int*   nodes = (const int*)d_in[6];
    const int*   nl    = (const int*)d_in[7];
    const int*   pl    = (const int*)d_in[8];
    const int*   ngl   = (const int*)d_in[9];

    const int ND = in_sizes[0];          // N * 64
    const int E  = in_sizes[1];
    const int B  = in_sizes[6];          // 8192
    const int P  = in_sizes[7];          // 262144

    char* w = (char*)d_ws;
    float*    e_cur  = (float*)w;     w += (size_t)ND * 4;
    float*    e_nxt  = (float*)w;     w += (size_t)ND * 4;
    float*    acc    = (float*)w;     w += (size_t)ND * 4;
    float*    n1f    = (float*)w;     w += (size_t)B * DIMS * 4;
    float*    n2f    = (float*)w;     w += (size_t)B * DIMS * 4;
    _Float16* n1h    = (_Float16*)w;  w += (size_t)B * DIMS * 2;
    _Float16* n2h    = (_Float16*)w;  w += (size_t)B * DIMS * 2;
    float*    rowsum = (float*)w;     w += (size_t)B * 4;
    float*    accum  = (float*)w;     w += 16;          // [0]=ssl [1]=bpr [2]=reg

    // zero rowsum + accumulators (contiguous)
    hipMemsetAsync(rowsum, 0, (size_t)B * 4 + 16, stream);

    const float* valsets[3] = { vals1, vals2, vals0 };
    const int ndBlocks = (ND + 255) / 256;

    for (int g = 0; g < 3; ++g) {
        init_copy_kernel<<<ndBlocks, 256, 0, stream>>>(emb, e_cur, acc, ND);
        for (int layer = 0; layer < 3; ++layer) {
            hipMemsetAsync(e_nxt, 0, (size_t)ND * 4, stream);
            spmm_kernel<<<(E + 15) / 16, 256, 0, stream>>>(e_cur, e_nxt, erow, ecol,
                                                           valsets[g], E);
            acc_add_kernel<<<ndBlocks, 256, 0, stream>>>(acc, e_nxt, ND);
            float* t = e_cur; e_cur = e_nxt; e_nxt = t;
        }
        if (g == 0) {
            gather_l1norm_kernel<<<B / 8, 256, 0, stream>>>(acc, nodes, n1f, n1h);
        } else if (g == 1) {
            gather_l1norm_kernel<<<B / 8, 256, 0, stream>>>(acc, nodes, n2f, n2h);
        }
        // g == 2: acc (= 4 * emb_all) stays resident for BPR
    }

    dim3 ggrid(B / (128 * NT), B / 16);
    ssl_gemm_kernel<<<ggrid, 256, 0, stream>>>(n1h, n2h, rowsum);
    ssl_pos_kernel<<<B / 8, 256, 0, stream>>>(n1f, n2f, rowsum, accum);
    bpr_kernel<<<P / 8, 256, 0, stream>>>(acc, emb, nl, pl, ngl, accum);
    finalize_kernel<<<1, 1, 0, stream>>>(accum, (float*)d_out,
                                         1.0f / (float)P, 0.5f / (float)B);
}
